// EdgeConvBlock_86861418594842
// MI455X (gfx1250) — matrix operand order repaired and checked by reading, not compiled, not run on hardware
//
#include <hip/hip_runtime.h>
#include <hip/hip_bf16.h>

typedef unsigned short u16;
typedef __attribute__((ext_vector_type(16))) __bf16 v16bf;
typedef __attribute__((ext_vector_type(8)))  float  v8f;

#define BATCH 8
#define CH    64
#define NPTS  4096
#define OCH   64
#define EDIM  128
#define KNN   20
#define CTILES 4                      // column tiles per barrier in knn
#define TPAD  (CTILES * 16 + 1)       // padded LDS row stride (bank-conflict-free scan)

// ---------------- workspace layout (bytes) ----------------
#define OFF_PTS   ((size_t)0)                         // (B,N,64) bf16   4 MB
#define OFF_SQ    ((size_t)4194304)                   // (B,N)   f32   128 KB
#define OFF_W1    ((size_t)4325376)                   // (64,64) bf16    8 KB
#define OFF_W2    ((size_t)4333568)                   // (64,64) bf16    8 KB
#define OFF_IDX   ((size_t)4341760)                   // (B,N,20) int  2.5 MB
#define OFF_G     ((size_t)6963200)                   // (B,N,64) f32    8 MB
#define OFF_H     ((size_t)15351808)                  // (B,N,64) f32    8 MB
#define OFF_PRE   ((size_t)23740416)                  // (B,64,N) f32    8 MB
#define OFF_MEAN  ((size_t)32129024)                  // (B*64)  f32
#define OFF_RSTD  ((size_t)32131072)
#define OFF_GAMMA ((size_t)32133120)
#define OFF_BETA  ((size_t)32135168)

__device__ __forceinline__ u16 f2bf(float f) {
  unsigned u = __float_as_uint(f);
  unsigned r = u + 0x7FFFu + ((u >> 16) & 1u);   // round-to-nearest-even
  return (u16)(r >> 16);
}

__device__ __forceinline__ v8f wmma_bf16(v16bf a, v16bf b, v8f c) {
  return __builtin_amdgcn_wmma_f32_16x16x32_bf16(false, a, false, b, (short)0, c, false, false);
}

// A-frag (16x32 bf16, MxK): lane<16 holds row M=lane, K in [c0,c0+16), c0=0;
// lane>=16 same row, c0=16.  kstep selects K-range [32*kstep, 32*kstep+32).
__device__ __forceinline__ v16bf load_afrag(const u16* rowbase, int lane, int kstep) {
  int c0 = kstep * 32 + ((lane & 16) ? 16 : 0);
  union { v16bf v; u16 u[16]; } f;
#pragma unroll
  for (int i = 0; i < 16; ++i) {
    f.u[i]     = rowbase[c0 + i];
  }
  return f.v;
}

// B-frag (32x16 bf16, KxN): lanes 0-15 hold K=[0,16), lanes 16-31 K=[16,32) for column N=lane&15.
__device__ __forceinline__ v16bf load_bfrag(const u16* colbase, int lane, int kstep) {
  int c0 = kstep * 32 + ((lane & 16) ? 16 : 0);
  return *(const v16bf*)(colbase + c0);
}

// ---------- prep: x (B,C,N) f32 -> pts (B,N,C) bf16 + squared norms ----------
__global__ void prep_pts_kernel(const float* __restrict__ x, u16* __restrict__ pts,
                                float* __restrict__ sq) {
  int tid = blockIdx.x * 256 + threadIdx.x;       // B*N = 32768
  if (tid >= BATCH * NPTS) return;
  int b = tid >> 12, n = tid & (NPTS - 1);
  float s = 0.f;
#pragma unroll 4
  for (int c = 0; c < CH; ++c) {
    float v = x[((size_t)(b * CH + c)) * NPTS + n];
    s += v * v;
    pts[(size_t)tid * CH + c] = f2bf(v);
  }
  sq[tid] = s;
}

// ---------- prep weights: W1 = conv_w[:, :64], W2' = conv_w[:, 64:] - W1 ----------
__global__ void prep_w_kernel(const float* __restrict__ cw, u16* __restrict__ w1,
                              u16* __restrict__ w2) {
  int tid = blockIdx.x * 256 + threadIdx.x;       // 64*64 = 4096
  if (tid >= OCH * CH) return;
  int o = tid >> 6, c = tid & 63;
  float a = cw[o * 2 * CH + c];
  float d = cw[o * 2 * CH + CH + c] - a;
  w1[tid] = f2bf(a);
  w2[tid] = f2bf(d);
}

// ---------- KNN: WMMA distance tiles + LDS-resident top-20 ----------
__global__ void knn_kernel(const u16* __restrict__ pts, const float* __restrict__ sq,
                           int* __restrict__ idx) {
  __shared__ float tile[16][TPAD];                // distance panel: 16 rows x 64 cols
  __shared__ float topd[16][KNN];                 // per-row sorted candidate distances
  __shared__ int   topi[16][KNN];                 // per-row candidate indices
  const int lane  = threadIdx.x;                  // 32 threads = 1 wave
  const int b     = blockIdx.y;
  const int rbase = blockIdx.x * 16;

  const u16* prow = pts + ((size_t)(b * NPTS + rbase + (lane & 15))) * CH;
  v16bf a0 = load_afrag(prow, lane, 0);
  v16bf a1 = load_afrag(prow, lane, 1);

  const int r0 = (lane >> 4) * 8;                 // C/D rows for this lane
  float sqa[8];
#pragma unroll
  for (int i = 0; i < 8; ++i) sqa[i] = sq[b * NPTS + rbase + r0 + i];

  if (lane < 16) {
#pragma unroll
    for (int k = 0; k < KNN; ++k) { topd[lane][k] = 3.4e38f; topi[lane][k] = 0; }
  }
  __syncthreads();

  float worst = 3.4e38f;                          // register copy of topd[row][KNN-1]
  const int grow = rbase + lane;                  // global row (valid for lane<16)

  for (int cb = 0; cb < NPTS / (16 * CTILES); ++cb) {   // 64 panels of 64 columns
#pragma unroll
    for (int s = 0; s < CTILES; ++s) {
      const int cbase = (cb * CTILES + s) * 16;
      const u16* crow = pts + ((size_t)(b * NPTS + cbase + (lane & 15))) * CH;
      if (s == 0 && cb + 1 < NPTS / (16 * CTILES))
        __builtin_prefetch(crow + CTILES * 16 * CH, 0, 1);  // global_prefetch_b8
      float sqb = sq[b * NPTS + cbase + (lane & 15)];

      v8f acc = {};
      acc = wmma_bf16(a0, load_bfrag(crow, lane, 0), acc);
      acc = wmma_bf16(a1, load_bfrag(crow, lane, 1), acc);
#pragma unroll
      for (int i = 0; i < 8; ++i)
        tile[r0 + i][s * 16 + (lane & 15)] = sqa[i] + sqb - 2.0f * acc[i];
    }
    __syncthreads();

    if (lane < 16) {
      const int colbase = cb * CTILES * 16;
      for (int j = 0; j < CTILES * 16; ++j) {
        float d = tile[lane][j];
        int col = colbase + j;
        if (d < worst && col != grow) {
          int pos = KNN - 1;                      // rare path: LDS insertion sort
          while (pos > 0 && topd[lane][pos - 1] > d) {
            topd[lane][pos] = topd[lane][pos - 1];
            topi[lane][pos] = topi[lane][pos - 1];
            --pos;
          }
          topd[lane][pos] = d;
          topi[lane][pos] = col;
          worst = topd[lane][KNN - 1];
        }
      }
    }
    __syncthreads();
  }

  if (lane < 16) {
#pragma unroll
    for (int k = 0; k < KNN; ++k)
      idx[((size_t)(b * NPTS + grow)) * KNN + k] = topi[lane][k];
  }
}

// ---------- G = pts*W1^T, H = pts*W2'^T  (B*N x 64 x 64, WMMA) ----------
__global__ void gemm_gh_kernel(const u16* __restrict__ pts, const u16* __restrict__ w1,
                               const u16* __restrict__ w2, float* __restrict__ G,
                               float* __restrict__ H) {
  const int lane  = threadIdx.x;
  const int b     = blockIdx.y;
  const int nbase = blockIdx.x * 16;

  const u16* prow = pts + ((size_t)(b * NPTS + nbase + (lane & 15))) * CH;
  v16bf a0 = load_afrag(prow, lane, 0);
  v16bf a1 = load_afrag(prow, lane, 1);

  const int r0 = (lane >> 4) * 8;
  const int oc = lane & 15;
#pragma unroll
  for (int ot = 0; ot < 4; ++ot) {
    const u16* wr1 = w1 + (ot * 16 + oc) * CH;
    const u16* wr2 = w2 + (ot * 16 + oc) * CH;
    v8f cg = {};
    cg = wmma_bf16(a0, load_bfrag(wr1, lane, 0), cg);
    cg = wmma_bf16(a1, load_bfrag(wr1, lane, 1), cg);
    v8f chh = {};
    chh = wmma_bf16(a0, load_bfrag(wr2, lane, 0), chh);
    chh = wmma_bf16(a1, load_bfrag(wr2, lane, 1), chh);
    int o = ot * 16 + oc;
#pragma unroll
    for (int i = 0; i < 8; ++i) {
      size_t off = ((size_t)(b * NPTS + nbase + r0 + i)) * OCH + o;
      G[off] = cg[i];
      H[off] = chh[i];
    }
  }
}

// ---------- out_pre[b,o,n] = max_k G[b,idx,o] + H[b,n,o] ----------
__global__ void gather_max_kernel(const float* __restrict__ G, const float* __restrict__ H,
                                  const int* __restrict__ idx, float* __restrict__ pre) {
  __shared__ int sidx[4][KNN];
  const int t = threadIdx.x;                      // 256 = 64 o x 4 points
  const int b = blockIdx.y;
  const int n0 = blockIdx.x * 4;
  if (t < 4 * KNN)
    sidx[t / KNN][t % KNN] = idx[((size_t)(b * NPTS + n0 + t / KNN)) * KNN + t % KNN];
  __syncthreads();
  const int o = t & 63, p = t >> 6;
  const int n = n0 + p;
  float m = -3.4e38f;
#pragma unroll
  for (int k = 0; k < KNN; ++k) {
    int j = sidx[p][k];
    m = fmaxf(m, G[((size_t)(b * NPTS + j)) * OCH + o]);
  }
  float h = H[((size_t)(b * NPTS + n)) * OCH + o];
  pre[((size_t)(b * OCH + o)) * NPTS + n] = m + h;
}

// ---------- per-(b,o) mean / rstd over N ----------
__global__ void stats_kernel(const float* __restrict__ pre, float* __restrict__ mean,
                             float* __restrict__ rstd) {
  __shared__ float ssum[256], ssq[256];
  const int bo = blockIdx.x;
  const int t  = threadIdx.x;
  float s = 0.f, q = 0.f;
  for (int i = t; i < NPTS; i += 256) {
    float v = pre[(size_t)bo * NPTS + i];
    s += v; q += v * v;
  }
  ssum[t] = s; ssq[t] = q;
  __syncthreads();
  for (int w = 128; w > 0; w >>= 1) {
    if (t < w) { ssum[t] += ssum[t + w]; ssq[t] += ssq[t + w]; }
    __syncthreads();
  }
  if (t == 0) {
    float mu  = ssum[0] * (1.0f / NPTS);
    float var = ssq[0] * (1.0f / NPTS) - mu * mu;
    mean[bo] = mu;
    rstd[bo] = rsqrtf(var + 1e-5f);
  }
}

// ---------- FiLM MLPs -> gamma, beta (tiny, one block) ----------
__global__ void film_kernel(const float* __restrict__ emb,
                            const float* __restrict__ gw1, const float* __restrict__ gb1,
                            const float* __restrict__ gw2, const float* __restrict__ gb2,
                            const float* __restrict__ bw1, const float* __restrict__ bb1,
                            const float* __restrict__ bw2, const float* __restrict__ bb2,
                            float* __restrict__ gamma, float* __restrict__ beta) {
  __shared__ float hg[BATCH][EDIM], hb[BATCH][EDIM];
  const int t = threadIdx.x;
  for (int id = t; id < BATCH * EDIM; id += 256) {
    int b = id >> 7, e = id & 127;
    float s1 = gb1[e], s2 = bb1[e];
    for (int j = 0; j < EDIM; ++j) {
      float x = emb[b * EDIM + j];
      s1 = fmaf(x, gw1[e * EDIM + j], s1);
      s2 = fmaf(x, bw1[e * EDIM + j], s2);
    }
    hg[b][e] = fmaxf(s1, 0.f);
    hb[b][e] = fmaxf(s2, 0.f);
  }
  __syncthreads();
  for (int id = t; id < BATCH * OCH; id += 256) {
    int b = id >> 6, o = id & 63;
    float s1 = gb2[o], s2 = bb2[o];
    for (int j = 0; j < EDIM; ++j) {
      s1 = fmaf(hg[b][j], gw2[o * EDIM + j], s1);
      s2 = fmaf(hb[b][j], bw2[o * EDIM + j], s2);
    }
    gamma[b * OCH + o] = 1.f + s1;
    beta[b * OCH + o]  = s2;
  }
}

// ---------- normalize + FiLM + ReLU epilogue (float4) ----------
__global__ void final_kernel(const float* __restrict__ pre, const float* __restrict__ mean,
                             const float* __restrict__ rstd, const float* __restrict__ gamma,
                             const float* __restrict__ beta, float* __restrict__ out) {
  int gid = blockIdx.x * 256 + threadIdx.x;       // float4 index
  int bo  = gid >> 10;                            // 4096/4 float4 per (b,o)
  float4 v = ((const float4*)pre)[gid];
  float mu = mean[bo], rs = rstd[bo], g = gamma[bo], be = beta[bo];
  v.x = fmaxf((v.x - mu) * rs * g + be, 0.f);
  v.y = fmaxf((v.y - mu) * rs * g + be, 0.f);
  v.z = fmaxf((v.z - mu) * rs * g + be, 0.f);
  v.w = fmaxf((v.w - mu) * rs * g + be, 0.f);
  ((float4*)out)[gid] = v;
}

extern "C" void kernel_launch(void* const* d_in, const int* in_sizes, int n_in,
                              void* d_out, int out_size, void* d_ws, size_t ws_size,
                              hipStream_t stream) {
  const float* x   = (const float*)d_in[0];
  const float* emb = (const float*)d_in[1];
  const float* cw  = (const float*)d_in[2];
  const float* gw1 = (const float*)d_in[3];
  const float* gb1 = (const float*)d_in[4];
  const float* gw2 = (const float*)d_in[5];
  const float* gb2 = (const float*)d_in[6];
  const float* bw1 = (const float*)d_in[7];
  const float* bb1 = (const float*)d_in[8];
  const float* bw2 = (const float*)d_in[9];
  const float* bb2 = (const float*)d_in[10];

  char* ws = (char*)d_ws;
  u16*   pts   = (u16*)(ws + OFF_PTS);
  float* sq    = (float*)(ws + OFF_SQ);
  u16*   w1    = (u16*)(ws + OFF_W1);
  u16*   w2    = (u16*)(ws + OFF_W2);
  int*   idx   = (int*)(ws + OFF_IDX);
  float* G     = (float*)(ws + OFF_G);
  float* H     = (float*)(ws + OFF_H);
  float* pre   = (float*)(ws + OFF_PRE);
  float* mean  = (float*)(ws + OFF_MEAN);
  float* rstd  = (float*)(ws + OFF_RSTD);
  float* gamma = (float*)(ws + OFF_GAMMA);
  float* beta  = (float*)(ws + OFF_BETA);
  float* out   = (float*)d_out;

  prep_pts_kernel<<<dim3((BATCH * NPTS + 255) / 256), dim3(256), 0, stream>>>(x, pts, sq);
  prep_w_kernel<<<dim3((OCH * CH + 255) / 256), dim3(256), 0, stream>>>(cw, w1, w2);
  film_kernel<<<dim3(1), dim3(256), 0, stream>>>(emb, gw1, gb1, gw2, gb2,
                                                 bw1, bb1, bw2, bb2, gamma, beta);
  knn_kernel<<<dim3(NPTS / 16, BATCH), dim3(32), 0, stream>>>(pts, sq, idx);
  gemm_gh_kernel<<<dim3(NPTS / 16, BATCH), dim3(32), 0, stream>>>(pts, w1, w2, G, H);
  gather_max_kernel<<<dim3(NPTS / 4, BATCH), dim3(256), 0, stream>>>(G, H, idx, pre);
  stats_kernel<<<dim3(BATCH * OCH), dim3(256), 0, stream>>>(pre, mean, rstd);
  final_kernel<<<dim3((BATCH * OCH * NPTS / 4) / 256), dim3(256), 0, stream>>>(
      pre, mean, rstd, gamma, beta, out);
}
